// TTRModelV2_43473658970332
// MI455X (gfx1250) — matrix operand order repaired and checked by reading, not compiled, not run on hardware
//
#include <hip/hip_runtime.h>
#include <hip/hip_bf16.h>

// ---------------- CDNA5 WMMA types ----------------
typedef __bf16  v16bf __attribute__((ext_vector_type(16)));
typedef __bf16  v8bf  __attribute__((ext_vector_type(8)));
typedef float   v8f   __attribute__((ext_vector_type(8)));
typedef unsigned short ushort_t;

#define DEVINL __device__ __forceinline__

DEVINL ushort_t f2bf_bits(float f) {
    union { float f; unsigned u; } c; c.f = f;
    unsigned u = c.u;
    return (ushort_t)((u + 0x7FFFu + ((u >> 16) & 1u)) >> 16);  // RNE
}

DEVINL v8f wmma_bf16(v16bf a, v16bf b, v8f c) {
    // D = A(16x32 bf16) * B(32x16 bf16) + C(16x16 f32)
    return __builtin_amdgcn_wmma_f32_16x16x32_bf16(false, a, false, b, (short)0, c, false, false);
}

// A fragment: lane holds row m, 16 contiguous K values [base,base+16)
DEVINL v16bf load_afrag(const ushort_t* p) {
    v8bf lo = *(const v8bf*)p;          // 16B aligned
    v8bf hi = *(const v8bf*)(p + 8);    // 16B aligned
    return __builtin_shufflevector(lo, hi, 0,1,2,3,4,5,6,7,8,9,10,11,12,13,14,15);
}
// B fragment from n-major Wt: lane holds col n, 16 contiguous K values
DEVINL v16bf load_bfrag(const ushort_t* p) {
    v8bf lo = *(const v8bf*)p;
    v8bf hi = *(const v8bf*)(p + 8);
    return __builtin_shufflevector(lo, hi, 0,1,2,3,4,5,6,7,8,9,10,11,12,13,14,15);
}

// float atomic max handling negatives (signed/unsigned int trick)
DEVINL void atomicMaxF(float* addr, float v) {
    if (v >= 0.f) atomicMax((int*)addr, __float_as_int(v));
    else          atomicMin((unsigned int*)addr, __float_as_uint(v));
}

// ---------------- preprocessing ----------------
// f32 -> bf16 elementwise
__global__ void k_cvt_bf16(const float* __restrict__ in, ushort_t* __restrict__ out, int n) {
    int i = blockIdx.x * 256 + threadIdx.x;
    if (i < n) out[i] = f2bf_bits(in[i]);
}

// W f32 [K,256] -> Wt bf16 [256,K]   (grid: (K/16, 16), block 256 = 16x16)
__global__ void k_transpose_bf16(const float* __restrict__ W, ushort_t* __restrict__ Wt, int K) {
    __shared__ float t[16][17];
    int k0 = blockIdx.x * 16, n0 = blockIdx.y * 16;
    int lx = threadIdx.x & 15, ly = threadIdx.x >> 4;
    t[ly][lx] = W[(size_t)(k0 + ly) * 256 + n0 + lx];
    __syncthreads();
    Wt[(size_t)(n0 + ly) * K + k0 + lx] = f2bf_bits(t[lx][ly]);
}

// =====================================================================
// Kernel A (encoders): out = relu(LN(Xb[M,K](bf16) @ Wt'[256,K] + bias))
// Block = 256 thr (8 waves), tile 16 x 256, LN via LDS.
// =====================================================================
__global__ void k_gemm_ln_relu(const ushort_t* __restrict__ Xb, int K,
                               const ushort_t* __restrict__ Wt,
                               const float* __restrict__ bias,
                               const float* __restrict__ lng,
                               const float* __restrict__ lnb,
                               float* __restrict__ outF,
                               ushort_t* __restrict__ outB, int M) {
    __shared__ float tile[16][264];
    const int tid = threadIdx.x, wave = tid >> 5, lane = tid & 31;
    const int hi = lane >> 4, m16 = lane & 15;
    const int row0 = blockIdx.x * 16;
    int rowA = row0 + m16; if (rowA >= M) rowA = M - 1;
    const ushort_t* __restrict__ arow = Xb + (size_t)rowA * (size_t)K;
    const ushort_t* __restrict__ w0 = Wt + (size_t)(wave * 32 + m16) * (size_t)K;
    const ushort_t* __restrict__ w1 = Wt + (size_t)(wave * 32 + 16 + m16) * (size_t)K;

    v8f c0 = {}; v8f c1 = {};
    const int nks = K >> 5;
    for (int ks = 0; ks < nks; ++ks) {
        v16bf a  = load_afrag(arow + ks * 32 + 16 * hi);
        v16bf b0 = load_bfrag(w0 + ks * 32 + 16 * hi);
        v16bf b1 = load_bfrag(w1 + ks * 32 + 16 * hi);
        c0 = wmma_bf16(a, b0, c0);
        c1 = wmma_bf16(a, b1, c1);
    }
#pragma unroll
    for (int r = 0; r < 8; ++r) {
        int rr = r + 8 * hi;
        int n0 = wave * 32 + m16;
        tile[rr][n0]      = c0[r] + bias[n0];
        tile[rr][n0 + 16] = c1[r] + bias[n0 + 16];
    }
    __syncthreads();

    for (int rr = 0; rr < 2; ++rr) {
        int row = wave * 2 + rr;
        int grow = row0 + row;
        float s = 0.f, s2 = 0.f;
#pragma unroll
        for (int i = 0; i < 8; ++i) { float v = tile[row][lane * 8 + i]; s += v; s2 += v * v; }
#pragma unroll
        for (int o = 16; o > 0; o >>= 1) { s += __shfl_xor(s, o); s2 += __shfl_xor(s2, o); }
        float mean = s * (1.f / 256.f);
        float rstd = rsqrtf(s2 * (1.f / 256.f) - mean * mean + 1e-5f);
        if (grow < M) {
#pragma unroll
            for (int i = 0; i < 8; ++i) {
                int col = lane * 8 + i;
                float v = (tile[row][col] - mean) * rstd * lng[col] + lnb[col];
                v = fmaxf(v, 0.f);
                if (outF) outF[(size_t)grow * 256 + col] = v;
                outB[(size_t)grow * 256 + col] = f2bf_bits(v);
            }
        }
    }
}

// =====================================================================
// Kernel B (main GEMM): out[M,256] = Xb(bf16)[M,256] @ W + bias
// Wt is bf16 n-major [256][256]. Block = 256 thr, tile 32 x 256:
// wave w -> rows [0,32) x cols [32w,32w+32) = 4 WMMA accumulators,
// A frags reused across 2 N-tiles, B frags across 2 M-tiles.
// =====================================================================
struct Mats4 { const ushort_t* W[4]; const float* b[4]; float* out[4]; };

__global__ void k_gemm256_bias(const ushort_t* __restrict__ Xb, Mats4 m, int M) {
    const int tid = threadIdx.x, wave = tid >> 5, lane = tid & 31;
    const int hi = lane >> 4, m16 = lane & 15;
    const int row0 = blockIdx.x * 32;
    const ushort_t* __restrict__ Wt  = m.W[blockIdx.y];
    const float* __restrict__ bias   = m.b[blockIdx.y];
    float* __restrict__ out          = m.out[blockIdx.y];

    int rA0 = row0 + m16;      if (rA0 >= M) rA0 = M - 1;
    int rA1 = row0 + 16 + m16; if (rA1 >= M) rA1 = M - 1;
    const ushort_t* __restrict__ a0p = Xb + (size_t)rA0 * 256;
    const ushort_t* __restrict__ a1p = Xb + (size_t)rA1 * 256;
    const ushort_t* __restrict__ w0  = Wt + (size_t)(wave * 32 + m16) * 256;
    const ushort_t* __restrict__ w1  = Wt + (size_t)(wave * 32 + 16 + m16) * 256;

    v8f c00 = {}, c01 = {}, c10 = {}, c11 = {};
#pragma unroll
    for (int ks = 0; ks < 8; ++ks) {
        const int ab = ks * 32 + 16 * hi;
        const int bb = ks * 32 + 16 * hi;
        v16bf a0 = load_afrag(a0p + ab);
        v16bf a1 = load_afrag(a1p + ab);
        v16bf b0 = load_bfrag(w0 + bb);
        v16bf b1 = load_bfrag(w1 + bb);
        c00 = wmma_bf16(a0, b0, c00);
        c10 = wmma_bf16(a1, b0, c10);
        c01 = wmma_bf16(a0, b1, c01);
        c11 = wmma_bf16(a1, b1, c11);
    }
    int n0 = wave * 32 + m16;
#pragma unroll
    for (int r = 0; r < 8; ++r) {
        int g0 = row0 + r + 8 * hi;
        int g1 = g0 + 16;
        if (g0 < M) {
            out[(size_t)g0 * 256 + n0]      = c00[r] + bias[n0];
            out[(size_t)g0 * 256 + n0 + 16] = c01[r] + bias[n0 + 16];
        }
        if (g1 < M) {
            out[(size_t)g1 * 256 + n0]      = c10[r] + bias[n0];
            out[(size_t)g1 * 256 + n0 + 16] = c11[r] + bias[n0 + 16];
        }
    }
}

// ---------------- attention (memory-bound, VALU + atomics) ----------------
__global__ void k_init_layer(float* attn, float* amax, float* denom, int N) {
    int i = blockIdx.x * 256 + threadIdx.x;
    if (i < N * 256) attn[i] = 0.f;
    if (i < N * 4) { amax[i] = -3.0e38f; denom[i] = 0.f; }
}

__global__ void k_attn1(const int* __restrict__ ei, const float* __restrict__ q,
                        const float* __restrict__ k, const float* __restrict__ ep,
                        float* __restrict__ alpha, float* __restrict__ amax, int E) {
    int w = (blockIdx.x * blockDim.x + threadIdx.x) >> 5;
    int lane = threadIdx.x & 31;
    if (w >= E) return;
    int src = ei[w], dst = ei[E + w];
    const float* qp  = q  + (size_t)dst * 256;
    const float* kp  = k  + (size_t)src * 256;
    const float* epp = ep + (size_t)w * 256;
    int d0 = lane * 8;          // 8 lanes per head, 8 dims per lane
    float s = 0.f;
#pragma unroll
    for (int i = 0; i < 8; ++i) { int d = d0 + i; s += qp[d] * (kp[d] + epp[d]); }
    s += __shfl_xor(s, 1); s += __shfl_xor(s, 2); s += __shfl_xor(s, 4);
    if ((lane & 7) == 0) {
        int h = lane >> 3;
        float a = s * 0.125f;   // 1/sqrt(64)
        alpha[(size_t)w * 4 + h] = a;
        atomicMaxF(&amax[(size_t)dst * 4 + h], a);
    }
}

__global__ void k_attn2(const int* __restrict__ ei, const float* __restrict__ alpha,
                        const float* __restrict__ amax, float* __restrict__ denom,
                        float* __restrict__ exb, int E) {
    int i = blockIdx.x * 256 + threadIdx.x;
    if (i >= E * 4) return;
    int e = i >> 2, h = i & 3;
    int dst = ei[E + e];
    float ex = __expf(alpha[i] - amax[(size_t)dst * 4 + h]);
    exb[i] = ex;
    atomicAdd(&denom[(size_t)dst * 4 + h], ex);
}

__global__ void k_attn3(const int* __restrict__ ei, const float* __restrict__ v,
                        const float* __restrict__ ep, const float* __restrict__ exb,
                        const float* __restrict__ denom, float* __restrict__ attn, int E) {
    int w = (blockIdx.x * blockDim.x + threadIdx.x) >> 5;
    int lane = threadIdx.x & 31;
    if (w >= E) return;
    int src = ei[w], dst = ei[E + w];
    int h = lane >> 3;
    float a = exb[(size_t)w * 4 + h] / (denom[(size_t)dst * 4 + h] + 1e-16f);
    const float* vp  = v  + (size_t)src * 256;
    const float* epp = ep + (size_t)w * 256;
    float* op = attn + (size_t)dst * 256;
    int d0 = lane * 8;
#pragma unroll
    for (int i = 0; i < 8; ++i) { int d = d0 + i; atomicAdd(&op[d], (vp[d] + epp[d]) * a); }
}

// h = relu(LN(h + attn + skip)); one wave per node row
__global__ void k_update(float* __restrict__ h, const float* __restrict__ attn,
                         const float* __restrict__ skip, const float* __restrict__ g,
                         const float* __restrict__ b, ushort_t* __restrict__ hB, int N) {
    int row = blockIdx.x * 8 + (threadIdx.x >> 5);
    int lane = threadIdx.x & 31;
    if (row >= N) return;
    float* hp = h + (size_t)row * 256;
    const float* ap = attn + (size_t)row * 256;
    const float* sp = skip + (size_t)row * 256;
    float vals[8]; float s = 0.f, s2 = 0.f;
#pragma unroll
    for (int i = 0; i < 8; ++i) {
        int d = lane * 8 + i;
        float t = hp[d] + ap[d] + sp[d];
        vals[i] = t; s += t; s2 += t * t;
    }
#pragma unroll
    for (int o = 16; o > 0; o >>= 1) { s += __shfl_xor(s, o); s2 += __shfl_xor(s2, o); }
    float mean = s * (1.f / 256.f);
    float rstd = rsqrtf(s2 * (1.f / 256.f) - mean * mean + 1e-5f);
#pragma unroll
    for (int i = 0; i < 8; ++i) {
        int d = lane * 8 + i;
        float t = fmaxf((vals[i] - mean) * rstd * g[d] + b[d], 0.f);
        hp[d] = t;
        hB[(size_t)row * 256 + d] = f2bf_bits(t);
    }
}

// ---------------- pooling ----------------
__global__ void k_init_pool(float* gsum, float* gmax) {
    gsum[threadIdx.x] = 0.f; gmax[threadIdx.x] = -3.0e38f;
}
__global__ void k_pool(const float* __restrict__ h, float* gsum, float* gmax, int N) {
    int d = threadIdx.x;
    int n0 = blockIdx.x * 256;
    int n1 = n0 + 256; if (n1 > N) n1 = N;
    float s = 0.f, mx = -3.0e38f;
    for (int n = n0; n < n1; ++n) {
        float v = h[(size_t)n * 256 + d];
        s += v; mx = fmaxf(mx, v);
    }
    atomicAdd(&gsum[d], s);
    atomicMaxF(&gmax[d], mx);
}

// ---------------- head MLPs (batch=1), single block ----------------
struct HeadP {
    const float *gsum, *gmax, *priv;
    const float *pl1W, *pl1b, *pn1g, *pn1b, *pl2W, *pl2b, *pn2g, *pn2b;
    const float *fl1W, *fl1b, *fn1g, *fn1b, *fl2W, *fl2b, *fn2g, *fn2b;
    const float *ql1W, *ql1b, *ql2W, *ql2b, *ql3W, *ql3b;
    const float *vl1W, *vl1b, *vl2W, *vl2b, *vl3W, *vl3b;
    float* out; float invN;
};

__device__ void block_ln_relu(float* buf, int n, const float* g, const float* b, float* red) {
    float s = 0.f, s2 = 0.f;
    for (int i = threadIdx.x; i < n; i += 256) { float v = buf[i]; s += v; s2 += v * v; }
    for (int o = 16; o > 0; o >>= 1) { s += __shfl_down(s, o); s2 += __shfl_down(s2, o); }
    int wave = threadIdx.x >> 5;
    if ((threadIdx.x & 31) == 0) { red[wave] = s; red[8 + wave] = s2; }
    __syncthreads();
    if (threadIdx.x == 0) {
        float ts = 0.f, ts2 = 0.f;
        for (int w = 0; w < 8; ++w) { ts += red[w]; ts2 += red[8 + w]; }
        float mean = ts / n;
        red[16] = mean; red[17] = rsqrtf(ts2 / n - mean * mean + 1e-5f);
    }
    __syncthreads();
    float mean = red[16], rstd = red[17];
    for (int i = threadIdx.x; i < n; i += 256)
        buf[i] = fmaxf((buf[i] - mean) * rstd * g[i] + b[i], 0.f);
    __syncthreads();
}

__global__ void k_head(HeadP P) {
    __shared__ float sP[128];
    __shared__ float sA[512];
    __shared__ float sB[768];
    __shared__ float sC[256];
    __shared__ float red[18];
    int t = threadIdx.x;
    if (t < 128) sP[t] = P.priv[t];
    sB[t]       = P.gsum[t] * P.invN;   // graph mean
    sB[256 + t] = P.gmax[t];            // graph max
    __syncthreads();
    for (int o = t; o < 512; o += 256) {
        float acc = P.pl1b[o];
        for (int i = 0; i < 128; ++i) acc += sP[i] * P.pl1W[(size_t)i * 512 + o];
        sA[o] = acc;
    }
    __syncthreads();
    block_ln_relu(sA, 512, P.pn1g, P.pn1b, red);
    {
        float acc = P.pl2b[t];
        for (int i = 0; i < 512; ++i) acc += sA[i] * P.pl2W[(size_t)i * 256 + t];
        sB[512 + t] = acc;
    }
    __syncthreads();
    block_ln_relu(sB + 512, 256, P.pn2g, P.pn2b, red);
    for (int o = t; o < 512; o += 256) {
        float acc = P.fl1b[o];
        for (int i = 0; i < 768; ++i) acc += sB[i] * P.fl1W[(size_t)i * 512 + o];
        sA[o] = acc;
    }
    __syncthreads();
    block_ln_relu(sA, 512, P.fn1g, P.fn1b, red);
    {
        float acc = P.fl2b[t];
        for (int i = 0; i < 512; ++i) acc += sA[i] * P.fl2W[(size_t)i * 256 + t];
        sC[t] = acc;
    }
    __syncthreads();
    block_ln_relu(sC, 256, P.fn2g, P.fn2b, red);
    // policy head
    { float a = P.ql1b[t]; for (int i = 0; i < 256; ++i) a += sC[i] * P.ql1W[(size_t)i * 256 + t]; sA[t] = fmaxf(a, 0.f); }
    __syncthreads();
    if (t < 128) { float a = P.ql2b[t]; for (int i = 0; i < 256; ++i) a += sA[i] * P.ql2W[(size_t)i * 128 + t]; sB[t] = fmaxf(a, 0.f); }
    __syncthreads();
    for (int o = t; o < 1000; o += 256) {
        float a = P.ql3b[o];
        for (int i = 0; i < 128; ++i) a += sB[i] * P.ql3W[(size_t)i * 1000 + o];
        P.out[o] = a;
    }
    __syncthreads();
    // value head
    { float a = P.vl1b[t]; for (int i = 0; i < 256; ++i) a += sC[i] * P.vl1W[(size_t)i * 256 + t]; sA[t] = fmaxf(a, 0.f); }
    __syncthreads();
    if (t < 128) { float a = P.vl2b[t]; for (int i = 0; i < 256; ++i) a += sA[i] * P.vl2W[(size_t)i * 128 + t]; sB[t] = fmaxf(a, 0.f); }
    __syncthreads();
    if (t == 0) {
        float a = P.vl3b[0];
        for (int i = 0; i < 128; ++i) a += sB[i] * P.vl3W[i];
        P.out[1000] = tanhf(a);
    }
}

// =====================================================================
extern "C" void kernel_launch(void* const* d_in, const int* in_sizes, int n_in,
                              void* d_out, int out_size, void* d_ws, size_t ws_size,
                              hipStream_t stream) {
    const float* x     = (const float*)d_in[0];
    const int*   ei    = (const int*)d_in[1];
    const float* eattr = (const float*)d_in[2];
    const int N = in_sizes[0] / 64;
    const int E = in_sizes[2] / 32;
    auto p = [&](int i) { return (const float*)d_in[i]; };

    char* ws = (char*)d_ws; size_t off = 0;
    auto carve = [&](size_t bytes) { char* r = ws + off; off += (bytes + 255) & ~(size_t)255; return r; };
    float*    hF    = (float*)carve((size_t)N * 256 * 4);
    ushort_t* hB    = (ushort_t*)carve((size_t)N * 256 * 2);
    ushort_t* ehB   = (ushort_t*)carve((size_t)E * 256 * 2);
    float*    qF    = (float*)carve((size_t)N * 256 * 4);
    float*    kF    = (float*)carve((size_t)N * 256 * 4);
    float*    vF    = (float*)carve((size_t)N * 256 * 4);
    float*    sF    = (float*)carve((size_t)N * 256 * 4);
    float*    epF   = (float*)carve((size_t)E * 256 * 4);
    float*    attn  = (float*)carve((size_t)N * 256 * 4);
    float*    alpha = (float*)carve((size_t)E * 4 * 4);
    float*    exb   = (float*)carve((size_t)E * 4 * 4);
    float*    amax  = (float*)carve((size_t)N * 4 * 4);
    float*    denom = (float*)carve((size_t)N * 4 * 4);
    float*    gsum  = (float*)carve(256 * 4);
    float*    gmax  = (float*)carve(256 * 4);
    ushort_t* xB    = (ushort_t*)carve((size_t)N * 64 * 2);   // bf16 node feats
    ushort_t* eaB   = (ushort_t*)carve((size_t)E * 32 * 2);   // bf16 edge attrs
    ushort_t* neWt  = (ushort_t*)carve((size_t)256 * 64 * 2); // node_enc Wt
    ushort_t* eeWt  = (ushort_t*)carve((size_t)256 * 32 * 2); // edge_enc Wt
    ushort_t* lWt[4][5];                                      // per layer q,k,v,skip,e
    for (int L = 0; L < 4; ++L)
        for (int j = 0; j < 5; ++j)
            lWt[L][j] = (ushort_t*)carve((size_t)256 * 256 * 2);
    (void)ws_size; (void)n_in; (void)out_size;

    dim3 blk(256);

    // ---- preprocessing: bf16 activations, transposed bf16 weights ----
    k_cvt_bf16<<<dim3((N * 64 + 255) / 256), blk, 0, stream>>>(x, xB, N * 64);
    k_cvt_bf16<<<dim3((E * 32 + 255) / 256), blk, 0, stream>>>(eattr, eaB, E * 32);
    k_transpose_bf16<<<dim3(64 / 16, 16), blk, 0, stream>>>(p(64), neWt, 64);   // node_enc.lin.W
    k_transpose_bf16<<<dim3(32 / 16, 16), blk, 0, stream>>>(p(4),  eeWt, 32);   // edge_enc.lin.W
    for (int L = 0; L < 4; ++L) {
        int GB = 16 + 12 * L;  // eW,eb,kW,kb,norm.b,norm.g,qW,qb,skipW,skipb,vW,vb
        const float* Ws[5] = { p(GB + 6), p(GB + 2), p(GB + 10), p(GB + 8), p(GB + 0) }; // q,k,v,skip,e
        for (int j = 0; j < 5; ++j)
            k_transpose_bf16<<<dim3(16, 16), blk, 0, stream>>>(Ws[j], lWt[L][j], 256);
    }

    // ---- encoders ----
    k_gemm_ln_relu<<<dim3((N + 15) / 16), blk, 0, stream>>>(xB, 64, neWt, p(65), p(67), p(66), hF, hB, N);
    k_gemm_ln_relu<<<dim3((E + 15) / 16), blk, 0, stream>>>(eaB, 32, eeWt, p(5), p(7), p(6), nullptr, ehB, E);

    // ---- GNN layers ----
    for (int L = 0; L < 4; ++L) {
        int GB = 16 + 12 * L;
        const float *eb = p(GB + 1), *kb = p(GB + 3), *nb = p(GB + 4), *ng = p(GB + 5);
        const float *qb = p(GB + 7), *sb = p(GB + 9), *vb = p(GB + 11);

        k_init_layer<<<dim3((N * 256 + 255) / 256), blk, 0, stream>>>(attn, amax, denom, N);

        Mats4 m4;
        m4.W[0] = lWt[L][0]; m4.W[1] = lWt[L][1]; m4.W[2] = lWt[L][2]; m4.W[3] = lWt[L][3];
        m4.b[0] = qb; m4.b[1] = kb; m4.b[2] = vb; m4.b[3] = sb;
        m4.out[0] = qF; m4.out[1] = kF; m4.out[2] = vF; m4.out[3] = sF;
        k_gemm256_bias<<<dim3((N + 31) / 32, 4), blk, 0, stream>>>(hB, m4, N);

        Mats4 me;
        for (int i = 0; i < 4; ++i) { me.W[i] = lWt[L][4]; me.b[i] = eb; me.out[i] = epF; }
        k_gemm256_bias<<<dim3((E + 31) / 32, 1), blk, 0, stream>>>(ehB, me, E);

        k_attn1<<<dim3((E + 7) / 8), blk, 0, stream>>>(ei, qF, kF, epF, alpha, amax, E);
        k_attn2<<<dim3((E * 4 + 255) / 256), blk, 0, stream>>>(ei, alpha, amax, denom, exb, E);
        k_attn3<<<dim3((E + 7) / 8), blk, 0, stream>>>(ei, vF, epF, exb, denom, attn, E);
        k_update<<<dim3((N + 7) / 8), blk, 0, stream>>>(hF, attn, sF, ng, nb, hB, N);
    }

    // ---- pooling + heads ----
    k_init_pool<<<1, blk, 0, stream>>>(gsum, gmax);
    k_pool<<<dim3((N + 255) / 256), blk, 0, stream>>>(hF, gsum, gmax, N);

    HeadP hp;
    hp.gsum = gsum; hp.gmax = gmax; hp.priv = p(3);
    hp.pl1W = p(74); hp.pl1b = p(75); hp.pn1g = p(79); hp.pn1b = p(78);
    hp.pl2W = p(76); hp.pl2b = p(77); hp.pn2g = p(81); hp.pn2b = p(80);
    hp.fl1W = p(8);  hp.fl1b = p(9);  hp.fn1g = p(13); hp.fn1b = p(12);
    hp.fl2W = p(10); hp.fl2b = p(11); hp.fn2g = p(15); hp.fn2b = p(14);
    hp.ql1W = p(68); hp.ql1b = p(69); hp.ql2W = p(70); hp.ql2b = p(71);
    hp.ql3W = p(72); hp.ql3b = p(73);
    hp.vl1W = p(82); hp.vl1b = p(83); hp.vl2W = p(84); hp.vl2b = p(85);
    hp.vl3W = p(86); hp.vl3b = p(87);
    hp.out = (float*)d_out; hp.invN = 1.0f / (float)N;
    k_head<<<1, blk, 0, stream>>>(hp);
}
